// BeatPatternExtractor_82222853915141
// MI455X (gfx1250) — 16-bit and 8-bit matrix operand order repaired and checked by reading, compiled, not run on hardware
//
#include <hip/hip_runtime.h>
#include <hip/hip_bf16.h>
#include <stdint.h>


typedef _Float16 half_t;
typedef __attribute__((ext_vector_type(16))) _Float16 v16h;
typedef __attribute__((ext_vector_type(8)))  float    v8f;
typedef __attribute__((ext_vector_type(8)))  int      v8i;

#define BATCH   32
#define CIN     256
#define LIN     5000
#define CO      128
#define KW      9
#define KTOT    (CIN*KW)
#define LOUT    2500
#define GPOOL   16
#define CHUNK   157
#define NW256   10
#define NTILE_L 157


__device__ inline v16h frag_a_f16(const half_t* A, int lda) {
  int lane = threadIdx.x & 31;
  int m  = lane & 15;
  int kb = (lane >> 4) * 16;
  v16h a;
#pragma unroll
  for (int e = 0; e < 16; ++e) {
    int K = kb + e;
    a[e] = A[m * lda + K];
  }
  return a;
}

__device__ inline v16h frag_b_f16(const half_t* B, int ldb) {
  int lane = threadIdx.x & 31;
  int n  = lane & 15;
  int kb = (lane >> 4) * 16;
  v16h b;
#pragma unroll
  for (int e = 0; e < 16; ++e) b[e] = B[(kb + e) * ldb + n];
  return b;
}

__device__ inline v8i frag_a_u8(const unsigned char* A, int lda) {
  int lane = threadIdx.x & 31;
  int m    = lane & 15;
  int off  = (lane >> 4) * 8;
  v8i a;
#pragma unroll
  for (int j = 0; j < 8; ++j) {
    int k0 = (j >> 1) * 16 + off + (j & 1) * 4;
    a[j] = *(const int*)(A + m * lda + k0);
  }
  return a;
}

__device__ inline v8i frag_b_u8_col(const unsigned char* Bt, int ldb) {
  int lane = threadIdx.x & 31;
  int n    = lane & 15;
  int base = (lane >> 4) * 16;
  v8i b;
#pragma unroll
  for (int j = 0; j < 8; ++j) {
    int k0 = (j >> 1) * 16 + base / 2 + (j & 1) * 4;
    b[j] = *(const int*)(Bt + n * ldb + k0);
  }
  return b;
}

__device__ inline void dcoord(int r, int& m, int& n) {
  int lane = threadIdx.x & 31;
  m = r + ((lane >> 4) << 3);
  n = lane & 15;
}

__global__ void k_quant_weights(const float* __restrict__ w, half_t* __restrict__ wqT) {
  __shared__ float red[256];
  int o = blockIdx.x, tid = threadIdx.x;
  float s = 0.f;
  for (int i = tid; i < KTOT; i += 256) s += fabsf(w[o * KTOT + i]);
  red[tid] = s; __syncthreads();
  for (int st = 128; st > 0; st >>= 1) { if (tid < st) red[tid] += red[tid + st]; __syncthreads(); }
  float alpha = red[0] * (1.0f / KTOT);
  for (int i = tid; i < KTOT; i += 256) {
    float v = w[o * KTOT + i];
    float q = (v > 0.f) ? alpha : (v < 0.f ? -alpha : 0.f);
    wqT[i * CO + o] = (half_t)q;
  }
}

__global__ void k_conv_proj_f16(const float* __restrict__ wq, const float* __restrict__ wk,
                                const float* __restrict__ wv, const float* __restrict__ wo,
                                half_t* __restrict__ out) {
  int idx = blockIdx.x * 256 + threadIdx.x;
  if (idx >= 4 * CO * CO) return;
  int sel = idx >> 14, r = idx & (CO * CO - 1);
  const float* src = (sel == 0) ? wq : (sel == 1) ? wk : (sel == 2) ? wv : wo;
  out[idx] = (half_t)src[r];
}

__global__ void __launch_bounds__(128)
k_conv_wmma(const float* __restrict__ x, const half_t* __restrict__ wqT,
            const float* __restrict__ convb, float* __restrict__ h) {
  __shared__ half_t As[16 * 36];
  __shared__ half_t Bs[32 * 132];
  int l0 = blockIdx.x * 16, b = blockIdx.y, tid = threadIdx.x;
  int wave = tid >> 5, n0 = wave * 32;
  v8f acc0 = {}, acc1 = {};

  for (int kk0 = 0; kk0 < KTOT; kk0 += 32) {
    if (kk0 + 32 < KTOT)
      __builtin_prefetch(wqT + (size_t)(kk0 + 32) * CO, 0, 0);
    for (int e = tid; e < 512; e += 128) {
      int m = e >> 5, kk = e & 31;
      int kt = kk0 + kk;
      int ci = kt / KW, kof = kt - ci * KW;
      int l = l0 + m;
      int il = 2 * l + kof - 4;
      float v = 0.f;
      if (l < LOUT && il >= 0 && il < LIN) v = x[((size_t)b * CIN + ci) * LIN + il];
      As[m * 36 + kk] = (half_t)v;
    }
    for (int e = tid; e < 4096; e += 128) {
      int r = e >> 7, c = e & 127;
      Bs[r * 132 + c] = wqT[(size_t)(kk0 + r) * CO + c];
    }
    __syncthreads();
    v16h a  = frag_a_f16(As, 36);
    v16h b0 = frag_b_f16(Bs + n0, 132);
    v16h b1 = frag_b_f16(Bs + n0 + 16, 132);
    acc0 = __builtin_amdgcn_wmma_f32_16x16x32_f16(false, a, false, b0, (short)0, acc0, false, false);
    acc1 = __builtin_amdgcn_wmma_f32_16x16x32_f16(false, a, false, b1, (short)0, acc1, false, false);
    __syncthreads();
  }
#pragma unroll
  for (int r = 0; r < 8; ++r) {
    int m, n; dcoord(r, m, n);
    int l = l0 + m;
    if (l < LOUT) {
      int o0 = n0 + n, o1 = n0 + 16 + n;
      h[((size_t)b * LOUT + l) * CO + o0] = acc0[r] + convb[o0];
      h[((size_t)b * LOUT + l) * CO + o1] = acc1[r] + convb[o1];
    }
  }
}

__global__ void k_bn_stats(const float* __restrict__ h, const float* __restrict__ gamma,
                           const float* __restrict__ beta, float* __restrict__ scale,
                           float* __restrict__ shift) {
  __shared__ float s1[256], s2[256];
  int c = blockIdx.x, tid = threadIdx.x;
  const int N = BATCH * LOUT;
  float a = 0.f, b = 0.f;
  for (int i = tid; i < N; i += 256) {
    float v = h[(size_t)i * CO + c];
    a += v; b += v * v;
  }
  s1[tid] = a; s2[tid] = b; __syncthreads();
  for (int st = 128; st > 0; st >>= 1) {
    if (tid < st) { s1[tid] += s1[tid + st]; s2[tid] += s2[tid + st]; }
    __syncthreads();
  }
  if (tid == 0) {
    float mu  = s1[0] / N;
    float var = s2[0] / N - mu * mu;
    float sc  = gamma[c] * rsqrtf(var + 1e-5f);
    scale[c] = sc;
    shift[c] = beta[c] - mu * sc;
  }
}

__global__ void k_bn_elu(float* __restrict__ h, half_t* __restrict__ h16,
                         const float* __restrict__ scale, const float* __restrict__ shift) {
  size_t idx = (size_t)blockIdx.x * 256 + threadIdx.x;
  if (idx >= (size_t)BATCH * LOUT * CO) return;
  int c = (int)(idx & (CO - 1));
  float v = h[idx] * scale[c] + shift[c];
  float e = (v > 0.f) ? v : (__expf(v) - 1.0f);
  h[idx] = e;
  h16[idx] = (half_t)e;
}

__global__ void __launch_bounds__(256)
k_qkv_wmma(const half_t* __restrict__ h16, const half_t* __restrict__ wproj,
           unsigned char* __restrict__ sq, unsigned char* __restrict__ sk,
           unsigned char* __restrict__ sv, float* __restrict__ Kp, float* __restrict__ Vp) {
  __shared__ half_t As[16 * 36];
  int l0 = blockIdx.x * 16, b = blockIdx.y, tid = threadIdx.x;
  int wave = tid >> 5, n0 = wave * 16;
  const half_t* Wq = wproj;
  const half_t* Wk = wproj + CO * CO;
  const half_t* Wv = wproj + 2 * CO * CO;
  v8f aq = {}, ak = {}, av = {};

  for (int kk0 = 0; kk0 < CO; kk0 += 32) {
    for (int e = tid; e < 512; e += 256) {
      int m = e >> 5, kk = e & 31;
      int l = l0 + m;
      As[m * 36 + kk] = (l < LOUT) ? h16[((size_t)b * LOUT + l) * CO + kk0 + kk] : (half_t)0.f;
    }
    __syncthreads();
    v16h a  = frag_a_f16(As, 36);
    v16h bq = frag_b_f16(Wq + kk0 * CO + n0, CO);
    v16h bk = frag_b_f16(Wk + kk0 * CO + n0, CO);
    v16h bv = frag_b_f16(Wv + kk0 * CO + n0, CO);
    aq = __builtin_amdgcn_wmma_f32_16x16x32_f16(false, a, false, bq, (short)0, aq, false, false);
    ak = __builtin_amdgcn_wmma_f32_16x16x32_f16(false, a, false, bk, (short)0, ak, false, false);
    av = __builtin_amdgcn_wmma_f32_16x16x32_f16(false, a, false, bv, (short)0, av, false, false);
    __syncthreads();
  }
#pragma unroll
  for (int r = 0; r < 8; ++r) {
    int m, n; dcoord(r, m, n);
    int l = l0 + m;
    if (l < LOUT) {
      int c = n0 + n;
      size_t idx = ((size_t)b * LOUT + l) * CO + c;
      sq[idx] = (aq[r] > 0.f) ? 1 : 0;
      sk[idx] = (ak[r] > 0.f) ? 1 : 0;
      sv[idx] = (av[r] > 0.f) ? 1 : 0;
      int g = l / CHUNK;
      atomicAdd(&Kp[((size_t)b * GPOOL + g) * CO + c], ak[r] * (1.0f / CHUNK));
      atomicAdd(&Vp[((size_t)b * GPOOL + g) * CO + c], av[r] * (1.0f / CHUNK));
    }
  }
}

__global__ void k_kv_outer(const float* __restrict__ Kp, const float* __restrict__ Vp,
                           unsigned char* __restrict__ kvT) {
  __shared__ unsigned char skg[GPOOL * CO], svg[GPOOL * CO];
  int b = blockIdx.x, tid = threadIdx.x;
  for (int i = tid; i < GPOOL * CO; i += 256) {
    skg[i] = (Kp[(size_t)b * GPOOL * CO + i] > 0.f) ? 1 : 0;
    svg[i] = (Vp[(size_t)b * GPOOL * CO + i] > 0.f) ? 1 : 0;
  }
  __syncthreads();
  for (int e = tid; e < CO * CO; e += 256) {
    int dd = e >> 7, c = e & 127;
    int s = 0;
#pragma unroll
    for (int g = 0; g < GPOOL; ++g) s += skg[g * CO + c] * svg[g * CO + dd];
    kvT[(size_t)b * CO * CO + e] = (unsigned char)s;
  }
}

__global__ void __launch_bounds__(256)
k_attn_wmma(const unsigned char* __restrict__ sqg, const unsigned char* __restrict__ skg,
            const unsigned char* __restrict__ svg, const unsigned char* __restrict__ kvg,
            float* __restrict__ aout) {
  extern __shared__ unsigned char lds[];
  unsigned char* sq_s = lds;
  unsigned char* sk_s = lds + 32768;
  unsigned char* svT  = lds + 65536;
  unsigned char* S_s  = lds + 98304;

  int wi = blockIdx.x, b = blockIdx.y, tid = threadIdx.x;
  int wave = tid >> 5;

  {
    unsigned long long qbase = (unsigned long long)(sqg + (size_t)b * LOUT * CO);
    unsigned long long kbase = (unsigned long long)(skg + (size_t)b * LOUT * CO);
#pragma unroll
    for (int it = 0; it < 8; ++it) {
      int i   = it * 256 + tid;
      int tok = i >> 3, seg = i & 7;
      int l   = wi * 256 + tok;
      int lc  = (l < LOUT) ? l : (LOUT - 1);
      unsigned goff = (unsigned)(lc * CO + seg * 16);
      unsigned dq   = (unsigned)(i * 16);
      unsigned dk   = (unsigned)(32768 + i * 16);
      asm volatile("global_load_async_to_lds_b128 %0, %1, %2"
                   :: "v"(dq), "v"(goff), "s"(qbase) : "memory");
      asm volatile("global_load_async_to_lds_b128 %0, %1, %2"
                   :: "v"(dk), "v"(goff), "s"(kbase) : "memory");
    }
  }
  for (int i = tid; i < 8192; i += 256) {
    int tok = i >> 5, w32 = i & 31;
    int l = wi * 256 + tok;
    unsigned int vv = 0;
    if (l < LOUT)
      vv = ((const unsigned int*)(svg + ((size_t)b * LOUT + l) * CO))[w32];
    int c0 = w32 * 4;
    svT[(c0 + 0) * 256 + tok] = (unsigned char)(vv & 255);
    svT[(c0 + 1) * 256 + tok] = (unsigned char)((vv >> 8) & 255);
    svT[(c0 + 2) * 256 + tok] = (unsigned char)((vv >> 16) & 255);
    svT[(c0 + 3) * 256 + tok] = (unsigned char)((vv >> 24) & 255);
  }
  asm volatile("s_wait_asynccnt 0x0" ::: "memory");
  __syncthreads();
  {
    int tstart = LOUT - wi * 256;
    if (tstart < 256) {
      int nz = (256 - tstart) * 32;
      for (int i = tid; i < nz; i += 256) {
        int idx = tstart * 32 + i;
        ((unsigned int*)sq_s)[idx] = 0;
        ((unsigned int*)sk_s)[idx] = 0;
      }
    }
  }
  __syncthreads();

  for (int t = wave; t < 256; t += 8) {
    int ti = t >> 4, tj = t & 15;
    v8i acc = {};
#pragma unroll
    for (int d0 = 0; d0 < CO; d0 += 64) {
      v8i a  = frag_a_u8(sq_s + (ti * 16) * CO + d0, CO);
      v8i bf = frag_b_u8_col(sk_s + (tj * 16) * CO + d0, CO);
      acc = __builtin_amdgcn_wmma_i32_16x16x64_iu8(false, a, false, bf, acc, false, false);
    }
#pragma unroll
    for (int r = 0; r < 8; ++r) {
      int m, n; dcoord(r, m, n);
      S_s[(ti * 16 + m) * 256 + tj * 16 + n] = (unsigned char)acc[r];
    }
  }
  __syncthreads();

  const unsigned char* kvTb = kvg + (size_t)b * CO * CO;
  for (int t = wave; t < 128; t += 8) {
    int ti = t >> 3, cj = t & 7;
    int row0 = ti * 16, c0 = cj * 16;
    v8i a256 = {}, a64 = {}, ag = {};
#pragma unroll
    for (int u0 = 0; u0 < 256; u0 += 64) {
      v8i a  = frag_a_u8(S_s + row0 * 256 + u0, 256);
      v8i bf = frag_b_u8_col(svT + c0 * 256 + u0, 256);
      a256 = __builtin_amdgcn_wmma_i32_16x16x64_iu8(false, a, false, bf, a256, false, false);
    }
    {
      int u0d = (ti >> 2) * 64;
      v8i a  = frag_a_u8(S_s + row0 * 256 + u0d, 256);
      v8i bf = frag_b_u8_col(svT + c0 * 256 + u0d, 256);
      a64 = __builtin_amdgcn_wmma_i32_16x16x64_iu8(false, a, false, bf, a64, false, false);
    }
#pragma unroll
    for (int d0 = 0; d0 < CO; d0 += 64) {
      v8i a  = frag_a_u8(sq_s + row0 * CO + d0, CO);
      v8i bf = frag_b_u8_col(kvTb + c0 * CO + d0, CO);
      ag = __builtin_amdgcn_wmma_i32_16x16x64_iu8(false, a, false, bf, ag, false, false);
    }
#pragma unroll
    for (int r = 0; r < 8; ++r) {
      int m, n; dcoord(r, m, n);
      int l = wi * 256 + row0 + m;
      if (l < LOUT)
        aout[((size_t)b * LOUT + l) * CO + c0 + n] =
            a256[r] * (1.0f / (256.0f * 3.0f)) +
            a64[r]  * (1.0f / (64.0f * 3.0f)) +
            ag[r]   * (1.0f / (16.0f * 3.0f));
    }
  }
}

__global__ void __launch_bounds__(128)
k_outproj_wmma(const float* __restrict__ aout, const half_t* __restrict__ wproj,
               const float* __restrict__ h, float* __restrict__ out) {
  __shared__ half_t As[16 * 132];
  int l0 = blockIdx.x * 16, b = blockIdx.y, tid = threadIdx.x;
  int wave = tid >> 5;
  const half_t* Wo = wproj + 3 * CO * CO;

  for (int e = tid; e < 16 * CO; e += 128) {
    int m = e >> 7, c = e & 127;
    int l = l0 + m;
    As[m * 132 + c] = (l < LOUT) ? (half_t)aout[((size_t)b * LOUT + l) * CO + c] : (half_t)0.f;
  }
  __syncthreads();

  v8f acc0 = {}, acc1 = {};
  int n0 = wave * 32;
#pragma unroll
  for (int kk0 = 0; kk0 < CO; kk0 += 32) {
    v16h a  = frag_a_f16(As + kk0, 132);
    v16h b0 = frag_b_f16(Wo + kk0 * CO + n0, CO);
    v16h b1 = frag_b_f16(Wo + kk0 * CO + n0 + 16, CO);
    acc0 = __builtin_amdgcn_wmma_f32_16x16x32_f16(false, a, false, b0, (short)0, acc0, false, false);
    acc1 = __builtin_amdgcn_wmma_f32_16x16x32_f16(false, a, false, b1, (short)0, acc1, false, false);
  }
#pragma unroll
  for (int r = 0; r < 8; ++r) {
    int m, n; dcoord(r, m, n);
    int l = l0 + m;
    if (l < LOUT) {
      int c0 = n0 + n, c1 = n0 + 16 + n;
      float h0 = h[((size_t)b * LOUT + l) * CO + c0];
      float h1 = h[((size_t)b * LOUT + l) * CO + c1];
      out[((size_t)b * CO + c0) * LOUT + l] = h0 + acc0[r];
      out[((size_t)b * CO + c1) * LOUT + l] = h1 + acc1[r];
    }
  }
}


extern "C" void kernel_launch(void* const* d_in, const int* in_sizes, int n_in,
                              void* d_out, int out_size, void* d_ws, size_t ws_size,
                              hipStream_t stream) {
  (void)in_sizes; (void)n_in; (void)out_size; (void)ws_size;
  const float* x      = (const float*)d_in[0];
  const float* conv_w = (const float*)d_in[1];
  const float* conv_b = (const float*)d_in[2];
  const float* gamma  = (const float*)d_in[3];
  const float* beta   = (const float*)d_in[4];
  const float* wq     = (const float*)d_in[5];
  const float* wk     = (const float*)d_in[6];
  const float* wv     = (const float*)d_in[7];
  const float* wo     = (const float*)d_in[8];
  float* out = (float*)d_out;

  char* ws = (char*)d_ws;
  half_t*        wqT   = (half_t*)(ws);
  half_t*        wproj = (half_t*)(ws + 589824);
  float*         scale = (float*)(ws + 720896);
  float*         shift = scale + 128;
  float*         hbuf  = (float*)(ws + 721920);
  half_t*        h16   = (half_t*)(ws + 41681920);
  unsigned char* sq    = (unsigned char*)(ws + 62161920);
  unsigned char* sk    = (unsigned char*)(ws + 72401920);
  unsigned char* sv    = (unsigned char*)(ws + 82641920);
  float*         Kp    = (float*)(ws + 92881920);
  float*         Vp    = (float*)(ws + 93144064);
  unsigned char* kvT   = (unsigned char*)(ws + 93406208);
  float*         aout  = (float*)(ws + 93930496);

  k_quant_weights<<<CO, 256, 0, stream>>>(conv_w, wqT);
  k_conv_proj_f16<<<(4 * CO * CO + 255) / 256, 256, 0, stream>>>(wq, wk, wv, wo, wproj);

  k_conv_wmma<<<dim3(NTILE_L, BATCH), 128, 0, stream>>>(x, wqT, conv_b, hbuf);

  k_bn_stats<<<CO, 256, 0, stream>>>(hbuf, gamma, beta, scale, shift);
  {
    size_t n = (size_t)BATCH * LOUT * CO;
    k_bn_elu<<<(unsigned)((n + 255) / 256), 256, 0, stream>>>(hbuf, h16, scale, shift);
  }

  hipMemsetAsync(Kp, 0, 2 * (size_t)BATCH * GPOOL * CO * sizeof(float), stream);
  k_qkv_wmma<<<dim3(NTILE_L, BATCH), 256, 0, stream>>>(h16, wproj, sq, sk, sv, Kp, Vp);

  k_kv_outer<<<BATCH, 256, 0, stream>>>(Kp, Vp, kvT);

  k_attn_wmma<<<dim3(NW256, BATCH), 256, 163840, stream>>>(sq, sk, sv, kvT, aout);

  k_outproj_wmma<<<dim3(NTILE_L, BATCH), 128, 0, stream>>>(aout, wproj, hbuf, out);
}
